// transformer_model_90048284328140
// MI455X (gfx1250) — matrix operand order repaired and checked by reading, not compiled, not run on hardware
//
#include <hip/hip_runtime.h>
#include <hip/hip_bf16.h>

// ---- problem constants (from reference) ----
#define L_   24
#define B_   32
#define S_   512
#define H_   1024
#define DIN  24576          // L_*H_
#define D1   900
#define D2   40
#define M_TOT (B_*S_)       // 16384 rows
#define K2PAD 928           // D1 padded to 29*32 for bf16 h1 buffer

// ---- GEMM1 tiling ----
#define BM 128
#define BN 64
#define BK 32
#define LDA (BK + 4)        // fp32 A tile row stride (36 floats, 16B-aligned rows)
#define LDB (BK + 8)        // bf16 B tile row stride (40 bf16, 16B-aligned rows)

typedef __attribute__((ext_vector_type(16))) __bf16 v16bf;
typedef __attribute__((ext_vector_type(8)))  float  v8f;
typedef __attribute__((ext_vector_type(4)))  int    i32x4_t;

static __device__ __forceinline__ __bf16 f2bf(float f) { return (__bf16)f; }

// ---- CDNA5 async global->LDS path (ASYNCcnt), with safe fallback ----
#if defined(__HIP_DEVICE_COMPILE__) && __has_builtin(__builtin_amdgcn_global_load_async_to_lds_b128)
#define HAS_ASYNC_LDS 1
#else
#define HAS_ASYNC_LDS 0
#endif

typedef __attribute__((address_space(1))) i32x4_t* as1_v4i;  // global 128-bit ptr
typedef __attribute__((address_space(3))) i32x4_t* as3_v4i;  // LDS 128-bit ptr

// copy 4 floats (16B) from global to LDS
static __device__ __forceinline__ void copy4f_to_lds(const float* g, float* l) {
#if HAS_ASYNC_LDS
  __builtin_amdgcn_global_load_async_to_lds_b128(
      (as1_v4i)(unsigned long long)g,
      (as3_v4i)(unsigned)(unsigned long long)l, 0, 0);
#else
  *(float4*)l = *(const float4*)g;
#endif
}

static __device__ __forceinline__ void wait_lds_async() {
#if HAS_ASYNC_LDS
#if __has_builtin(__builtin_amdgcn_s_wait_asynccnt)
  __builtin_amdgcn_s_wait_asynccnt(0);
#else
  asm volatile("s_wait_asynccnt 0x0" ::: "memory");
#endif
#endif
}

// ============================================================================
// Kernel 1: h1[m, n] = relu( X @ W1 + b1 ),  X[m, k=l*1024+h] = hs[l, b, s, h]
// 8 waves/block, block tile 128x64, wave tile 32x32 (2x2 WMMA 16x16x32 bf16).
// A staged via GLOBAL_LOAD_ASYNC_TO_LDS_B128 (fp32, converted at frag build),
// B staged sync with fp32->bf16 + transpose. Double-buffered LDS, 1 barrier/iter.
// ============================================================================
__global__ __launch_bounds__(256)
void gemm1_wmma(const float* __restrict__ hs, const float* __restrict__ W1,
                const float* __restrict__ b1, __bf16* __restrict__ h1)
{
  __shared__ float  lsA[2][BM][LDA];   // raw fp32 A tiles (m, k)
  __shared__ __bf16 lsB[2][BN][LDB];   // bf16 W1 tiles, transposed (n, k)

  const int tid  = threadIdx.x;
  const int lane = tid & 31;
  const int wave = tid >> 5;
  const int wm   = wave & 3;           // wave M offset: wm*32
  const int wn   = wave >> 2;          // wave N offset: wn*32
  const int n0   = blockIdx.x * BN;
  const int m0   = blockIdx.y * BM;

  // staging assignments
  const int arow  = tid >> 1;          // 0..127
  const int ahalf = (tid & 1) * 16;    // 0 or 16 within BK
  const int brow  = tid >> 3;          // 0..31  (k within tile)
  const int bcol  = (tid & 7) * 8;     // 0..56  (n within tile)

  const long long lstride = (long long)B_ * S_ * H_;   // layer stride in hs
  const bool bn_full = (n0 + BN <= D1);                // uniform over block

  v8f acc[2][2] = {};

  auto stage_A = [&](int kk, int pb) {   // async: 4 x b128 per thread
    const int l  = kk >> 10;
    const int h0 = (kk & 1023) + ahalf;  // K-step lives inside one layer chunk
    const float* src = hs + (long long)l * lstride + (long long)(m0 + arow) * H_ + h0;
    float* dst = &lsA[pb][arow][ahalf];
    copy4f_to_lds(src + 0,  dst + 0);
    copy4f_to_lds(src + 4,  dst + 4);
    copy4f_to_lds(src + 8,  dst + 8);
    copy4f_to_lds(src + 12, dst + 12);
  };

  auto stage_B = [&](int kk, int pb) {   // sync: cvt + transpose into LDS
    const float* src = W1 + (long long)(kk + brow) * D1 + (n0 + bcol);
    if (bn_full) {                       // uniform branch: no lane divergence
      const float4 f0 = ((const float4*)src)[0];
      const float4 f1 = ((const float4*)src)[1];
      lsB[pb][bcol + 0][brow] = f2bf(f0.x); lsB[pb][bcol + 1][brow] = f2bf(f0.y);
      lsB[pb][bcol + 2][brow] = f2bf(f0.z); lsB[pb][bcol + 3][brow] = f2bf(f0.w);
      lsB[pb][bcol + 4][brow] = f2bf(f1.x); lsB[pb][bcol + 5][brow] = f2bf(f1.y);
      lsB[pb][bcol + 6][brow] = f2bf(f1.z); lsB[pb][bcol + 7][brow] = f2bf(f1.w);
    } else {                             // only the last N-tile (n0 = 896)
      #pragma unroll
      for (int j = 0; j < 8; ++j) {
        float v = (n0 + bcol + j < D1) ? src[j] : 0.f;
        lsB[pb][bcol + j][brow] = f2bf(v);
      }
    }
  };

  const int lm  = lane & 15;
  const int hiA = (lane >> 4) ? 8  : 0;   // A frag: lanes 16-31 hold K+8
  const int klo = (lane >> 4) ? 16 : 0;   // unused; B frag indexes with hiA, same k order as A

  // prologue: stage first tile
  stage_A(0, 0);
  stage_B(0, 0);

  int pb = 0;
  for (int kk = 0; kk < DIN; kk += BK, pb ^= 1) {
    wait_lds_async();      // my async writes into lsA[pb] landed
    __syncthreads();       // everyone's staging of pb done; prev reads of pb^1 done

    if (kk + BK < DIN) {   // stage next tile into the other buffer (overlaps WMMA)
      stage_A(kk + BK, pb ^ 1);
      stage_B(kk + BK, pb ^ 1);
    }
    if (kk + 2 * BK < DIN) {  // L2 prefetch two steps ahead (global_prefetch_b8)
      const int kn = kk + 2 * BK;
      __builtin_prefetch(hs + (long long)(kn >> 10) * lstride +
                         (long long)(m0 + arow) * H_ + ((kn & 1023) + ahalf), 0, 1);
    }

    // ---- build fragments per documented VGPR layouts ----
    v16bf afrag[2], bfrag[2];
    #pragma unroll
    for (int mi = 0; mi < 2; ++mi) {
      const float* ar = &lsA[pb][wm * 32 + mi * 16 + lm][0];
      #pragma unroll
      for (int e = 0; e < 8; ++e) {       // VGPRs 0..3: K=hiA+e ; 4..7: K=16+hiA+e
        afrag[mi][e]     = f2bf(ar[hiA + e]);
        afrag[mi][8 + e] = f2bf(ar[16 + hiA + e]);
      }
    }
    #pragma unroll
    for (int ni = 0; ni < 2; ++ni) {
      const __bf16* br = &lsB[pb][wn * 32 + ni * 16 + lm][0];
      #pragma unroll
      for (int e = 0; e < 16; ++e)        // e<8: K=hiA+e ; e>=8: K=16+hiA+(e-8)
        bfrag[ni][e] = br[(e < 8) ? (hiA + e) : (16 + hiA + (e - 8))];
    }

    #pragma unroll
    for (int mi = 0; mi < 2; ++mi)
      #pragma unroll
      for (int ni = 0; ni < 2; ++ni)
        acc[mi][ni] = __builtin_amdgcn_wmma_f32_16x16x32_bf16(
            false, afrag[mi], false, bfrag[ni],
            (short)0, acc[mi][ni], false, false);
  }

  // ---- epilogue: bias + relu, store bf16 into h1[M_TOT][K2PAD] ----
  const int mhi = (lane >> 4) ? 8 : 0;   // C layout: VGPR i -> M = i (+8 for hi lanes)
  #pragma unroll
  for (int mi = 0; mi < 2; ++mi) {
    #pragma unroll
    for (int ni = 0; ni < 2; ++ni) {
      const int col = n0 + wn * 32 + ni * 16 + lm;
      if (col >= K2PAD) continue;
      const float bias = (col < D1) ? b1[col] : 0.f;
      #pragma unroll
      for (int i = 0; i < 8; ++i) {
        const int row = m0 + wm * 32 + mi * 16 + mhi + i;
        float v = (col < D1) ? fmaxf(acc[mi][ni][i] + bias, 0.f) : 0.f;
        h1[(long long)row * K2PAD + col] = f2bf(v);
      }
    }
  }
}

// ============================================================================
// Kernel 2: per-row 900 -> 40 (relu) -> 1 (sigmoid). 4 rows/block, 64 thr/row.
// ============================================================================
__global__ __launch_bounds__(256)
void mlp_tail(const __bf16* __restrict__ h1, const float* __restrict__ W2,
              const float* __restrict__ b2, const float* __restrict__ W3,
              const float* __restrict__ b3, float* __restrict__ logits)
{
  __shared__ float h2s[4][D2];
  const int rloc = threadIdx.x >> 6;     // 0..3
  const int t    = threadIdx.x & 63;
  const int m    = blockIdx.x * 4 + rloc;

  if (t < D2) {
    const __bf16* xr = h1 + (long long)m * K2PAD;
    float s = 0.f;
    for (int k = 0; k < D1; ++k)
      s += (float)xr[k] * W2[k * D2 + t];
    h2s[rloc][t] = fmaxf(s + b2[t], 0.f);
  }
  __syncthreads();
  if (t == 0) {
    float s = 0.f;
    #pragma unroll
    for (int n = 0; n < D2; ++n) s += h2s[rloc][n] * W3[n];
    s += b3[0];
    logits[m] = 1.f / (1.f + __expf(-s));
  }
}

// ============================================================================
// Kernel 3: ragged gather by token-start index (0 == "no word starts here")
// ============================================================================
__global__ __launch_bounds__(256)
void gather_starts(const float* __restrict__ logits,
                   const int* __restrict__ starts,
                   float* __restrict__ out)
{
  const int i = blockIdx.x * blockDim.x + threadIdx.x;
  if (i >= M_TOT) return;
  const int b   = i >> 9;                 // i / 512
  const int idx = starts[i];
  int c = idx < 0 ? 0 : (idx > S_ - 1 ? S_ - 1 : idx);
  out[i] = (idx != 0) ? logits[(b << 9) + c] : 0.f;
}

// ============================================================================
extern "C" void kernel_launch(void* const* d_in, const int* in_sizes, int n_in,
                              void* d_out, int out_size, void* d_ws, size_t ws_size,
                              hipStream_t stream) {
  const float* hs     = (const float*)d_in[0];
  const float* W1     = (const float*)d_in[1];
  const float* b1     = (const float*)d_in[2];
  const float* W2     = (const float*)d_in[3];
  const float* b2     = (const float*)d_in[4];
  const float* W3     = (const float*)d_in[5];
  const float* b3     = (const float*)d_in[6];
  const int*   starts = (const int*)d_in[7];

  __bf16* h1     = (__bf16*)d_ws;
  float*  logits = (float*)((char*)d_ws + (size_t)M_TOT * K2PAD * sizeof(__bf16));

  dim3 g1((D1 + BN - 1) / BN, M_TOT / BM);       // 15 x 128 tiles
  gemm1_wmma<<<g1, 256, 0, stream>>>(hs, W1, b1, h1);

  mlp_tail<<<M_TOT / 4, 256, 0, stream>>>(h1, W2, b2, W3, b3, logits);

  gather_starts<<<(M_TOT + 255) / 256, 256, 0, stream>>>(logits, starts, (float*)d_out);
}
